// CA_76922864272011
// MI455X (gfx1250) — matrix operand order repaired and checked by reading, not compiled, not run on hardware
//
#include <hip/hip_runtime.h>
#include <hip/hip_bf16.h>
#include <math.h>

// Problem constants (from reference)
#define BB 2
#define L1 512
#define L2 512
#define DD 256
#define NG 16
#define DG 16

typedef float v8f  __attribute__((ext_vector_type(8)));
typedef float v2f  __attribute__((ext_vector_type(2)));
typedef _Float16 v16h __attribute__((ext_vector_type(16)));

// ---------------------------------------------------------------------------
// Kernel 1: QKV projections via WMMA.
//   mat 0: Q = x_target @ Wq   (rows = B*L1 = 1024)
//   mat 1: K = x_source @ Wk   (rows = B*L2 = 1024)
//   mat 2: V = x_source @ Wv
// One wave computes one 16x16 output tile. 3 * (64*16) = 3072 tiles,
// 8 waves/block -> 384 blocks.
// ---------------------------------------------------------------------------
__global__ __launch_bounds__(256) void qkv_wmma_kernel(
    const float* __restrict__ xs, const float* __restrict__ xt,
    const float* __restrict__ Wq, const float* __restrict__ Wk,
    const float* __restrict__ Wv,
    float* __restrict__ Qd, float* __restrict__ Kd, float* __restrict__ Vd)
{
    const int wave = (blockIdx.x * blockDim.x + threadIdx.x) >> 5;
    const int lane = threadIdx.x & 31;
    const int mat  = wave >> 10;        // 0..2
    const int t    = wave & 1023;       // tile within matrix
    const int rowTile = t >> 4;         // 0..63
    const int colTile = t & 15;         // 0..15

    const float* __restrict__ X = (mat == 0) ? xt : xs;
    const float* __restrict__ W = (mat == 0) ? Wq : ((mat == 1) ? Wk : Wv);
    float*       __restrict__ O = (mat == 0) ? Qd : ((mat == 1) ? Kd : Vd);

    const int row0 = rowTile * 16;
    const int col0 = colTile * 16;
    const int hl = lane >> 4;           // lane half: 0 or 1
    const int ll = lane & 15;

    v8f c = {};

#if __has_builtin(__builtin_amdgcn_wmma_f32_16x16x4_f32)
    // Exact-fp32 path: V_WMMA_F32_16X16X4_F32, K stepped by 4.
    // A (16x4, 2 VGPR): lanes0-15 K={0,1}, lanes16-31 K={2,3}; M = lane%16.
    // B (4x16, 2 VGPR): mirrored; N = lane%16.
    #pragma unroll 4
    for (int k0 = 0; k0 < DD; k0 += 4) {
        v2f a, b;
        const float* arow = X + (row0 + ll) * DD + k0 + 2 * hl;
        a.x = arow[0];
        a.y = arow[1];
        const float* bcol = W + (k0 + 2 * hl) * DD + col0 + ll;
        b.x = bcol[0];
        b.y = bcol[DD];
        c = __builtin_amdgcn_wmma_f32_16x16x4_f32(
                false, a, false, b, (short)0, c, false, false);
    }
#else
    // Fallback: f16 inputs, f32 accumulate (codegen-confirmed builtin).
    // A (16x32 f16): VGPR v -> K = 16*hl + 2v + {0,1}; M = lane%16.
    // B (32x16 f16): VGPR v -> K = 16*hl + 2v + {0,1}; N = lane%16.
    #pragma unroll
    for (int k0 = 0; k0 < DD; k0 += 32) {
        v16h a, b;
        const float* arow = X + (row0 + ll) * DD + k0 + 16 * hl;
        #pragma unroll
        for (int v = 0; v < 4; ++v) {
            a[2 * v]     = (_Float16)arow[2 * v];
            a[2 * v + 1] = (_Float16)arow[2 * v + 1];
        }
        #pragma unroll
        for (int v = 0; v < 4; ++v) {
            a[2 * (v + 4)]     = (_Float16)arow[8 + 2 * v];
            a[2 * (v + 4) + 1] = (_Float16)arow[8 + 2 * v + 1];
        }
        const float* bcol = W + (k0 + 16 * hl) * DD + col0 + ll;
        #pragma unroll
        for (int v = 0; v < 8; ++v) {
            b[2 * v]     = (_Float16)bcol[(2 * v) * DD];
            b[2 * v + 1] = (_Float16)bcol[(2 * v + 1) * DD];
        }
        c = __builtin_amdgcn_wmma_f32_16x16x32_f16(
                false, a, false, b, (short)0, c, false, false);
    }
#endif

    // C/D layout: VGPR j -> row j (lanes 0-15) / row j+8 (lanes 16-31), col = lane%16
    float* orow = O + (row0 + 8 * hl) * DD + col0 + ll;
    #pragma unroll
    for (int j = 0; j < 8; ++j) orow[j * DD] = c[j];
}

// ---------------------------------------------------------------------------
// Kernel 2: fused score + softmax + attn*V.
// One 256-thread block per (b, l) row (grid = B*L1 = 1024 blocks).
// ---------------------------------------------------------------------------
__global__ __launch_bounds__(256) void attn_fused_kernel(
    const float* __restrict__ Qd, const float* __restrict__ Kd,
    const float* __restrict__ Vd,
    const float* __restrict__ add_bias, const float* __restrict__ mul_bias,
    const float* __restrict__ w_mlp, const float* __restrict__ b_mlp,
    float* __restrict__ out)
{
    const int n   = blockIdx.x;      // n = b*L1 + l
    const int b   = n >> 9;          // L1 == 512
    const int tid = threadIdx.x;     // 0..255

    __shared__ float sQ[DD];
    __shared__ float sL[NG * (L2 + 1)];   // stride 513: bank-conflict-free

    // Q row for this (b, l)
    sQ[tid] = Qd[n * DD + tid];

    float wreg[DG];
    #pragma unroll
    for (int d = 0; d < DG; ++d) wreg[d] = w_mlp[d];
    const float bmlp = b_mlp[0];

    __syncthreads();

    // ---- Phase A: logits for all (g, m). Each thread covers m = tid, tid+256.
    #pragma unroll
    for (int mi = 0; mi < 2; ++mi) {
        const int m = (mi << 8) + tid;
        const float4* Kr =
            reinterpret_cast<const float4*>(Kd + (size_t)(b * L2 + m) * DD);
        #pragma unroll
        for (int g = 0; g < NG; ++g) {
            float s = 0.0f;
            #pragma unroll
            for (int d4 = 0; d4 < 4; ++d4) {
                const float4 kv = Kr[g * 4 + d4];
                const int d0 = g * DG + d4 * 4;
                s = fmaf(fmaxf(sQ[d0 + 0] - kv.x, 0.0f), wreg[d4 * 4 + 0], s);
                s = fmaf(fmaxf(sQ[d0 + 1] - kv.y, 0.0f), wreg[d4 * 4 + 1], s);
                s = fmaf(fmaxf(sQ[d0 + 2] - kv.z, 0.0f), wreg[d4 * 4 + 2], s);
                s = fmaf(fmaxf(sQ[d0 + 3] - kv.w, 0.0f), wreg[d4 * 4 + 3], s);
            }
            const float score = fmaxf(s + bmlp, 0.0f);
            const int bidx = (n * NG + g) * L2 + m;          // [B,L1,NG,L2]
            sL[g * (L2 + 1) + m] = fmaf(score, mul_bias[bidx], add_bias[bidx]);
        }
    }
    __syncthreads();

    // ---- Phase B: softmax over m (L2=512) per group.
    // 16 threads per group (consecutive lanes); 32 m-values per thread.
    {
        const int g = tid >> 4;
        const int i = tid & 15;
        float* row = sL + g * (L2 + 1);

        float vmax = -INFINITY;
        #pragma unroll 4
        for (int k = 0; k < 32; ++k) vmax = fmaxf(vmax, row[i + 16 * k]);
        #pragma unroll
        for (int off = 8; off >= 1; off >>= 1)
            vmax = fmaxf(vmax, __shfl_xor(vmax, off, 16));

        float lsum = 0.0f;
        #pragma unroll 4
        for (int k = 0; k < 32; ++k) {
            const int idx = i + 16 * k;
            const float e = __expf(row[idx] - vmax);
            row[idx] = e;
            lsum += e;
        }
        #pragma unroll
        for (int off = 8; off >= 1; off >>= 1)
            lsum += __shfl_xor(lsum, off, 16);

        const float rinv = 1.0f / lsum;
        #pragma unroll 4
        for (int k = 0; k < 32; ++k) row[i + 16 * k] *= rinv;
    }
    __syncthreads();

    // ---- Phase C: out[b,l,d,g] = sum_m attn[g][m] * V[b,m,d*16+g].
    // tid == d*16+g maps exactly onto V's innermost dim -> coalesced.
    {
        const float* Vb = Vd + (size_t)b * L2 * DD + tid;
        const float* Ag = sL + (tid & 15) * (L2 + 1);
        float acc = 0.0f;
        #pragma unroll 4
        for (int m = 0; m < L2; ++m)
            acc = fmaf(Ag[m], Vb[(size_t)m * DD], acc);
        out[n * DD + tid] = acc;
    }
}

extern "C" void kernel_launch(void* const* d_in, const int* in_sizes, int n_in,
                              void* d_out, int out_size, void* d_ws, size_t ws_size,
                              hipStream_t stream) {
    const float* xs   = (const float*)d_in[0];  // x_source [B,L2,D]
    const float* xt   = (const float*)d_in[1];  // x_target [B,L1,D]
    const float* addb = (const float*)d_in[2];  // [B,L1,NG,L2]
    const float* mulb = (const float*)d_in[3];  // [B,L1,NG,L2]
    const float* Wq   = (const float*)d_in[4];
    const float* Wk   = (const float*)d_in[5];
    const float* Wv   = (const float*)d_in[6];
    const float* wm   = (const float*)d_in[7];  // [DG]
    const float* bm   = (const float*)d_in[8];  // scalar
    float* out = (float*)d_out;

    float* Qd = (float*)d_ws;                       // [B*L1, D]
    float* Kd = Qd + (size_t)BB * L1 * DD;          // [B*L2, D]
    float* Vd = Kd + (size_t)BB * L2 * DD;          // [B*L2, D]

    // 3 matrices * 1024 tiles each = 3072 waves; 8 waves per 256-thread block.
    qkv_wmma_kernel<<<384, 256, 0, stream>>>(xs, xt, Wq, Wk, Wv, Qd, Kd, Vd);
    attn_fused_kernel<<<BB * L1, 256, 0, stream>>>(Qd, Kd, Vd, addb, mulb,
                                                   wm, bm, out);
}
